// SE_Residual_50173807952094
// MI455X (gfx1250) — matrix operand order repaired and checked by reading, not compiled, not run on hardware
//
#include <hip/hip_runtime.h>
#include <cstdint>
#include <cstddef>

// ---------------------------------------------------------------------------
// SE block (per-region-position squeeze/excite) for MI455X / gfx1250.
// Rows grouped by region position j (weights shared within j); two bf16 WMMA
// GEMMs per 16-row tile; fused bias/relu/sigmoid/residual. Memory-bound on
// x in/out (~226MB @ 23.3TB/s); bf16 weights (38.8MB) stay L2-resident.
// ---------------------------------------------------------------------------

typedef __attribute__((ext_vector_type(16))) __bf16 v16bf_t;
typedef __attribute__((ext_vector_type(8)))  float  v8f_t;

union Frag16 { v16bf_t v; uint4 u[2]; };

// Native bf16 converts: let clang lower fptrunc (v_cvt_*bf16_f32 on gfx1250)
__device__ __forceinline__ unsigned short f2bf(float f) {
  return __builtin_bit_cast(unsigned short, (__bf16)f);
}
__device__ __forceinline__ unsigned int pack2bf(float a, float b) {
  unsigned int lo = f2bf(a);
  unsigned int hi = f2bf(b);
  return lo | (hi << 16);
}

constexpr int CH    = 2048;   // channels
constexpr int HID   = 128;    // hidden (reduction 16)
constexpr int RP    = 37;     // max regions per sample
constexpr int BSAMP = 512;    // samples

// --------------------------- weight f32 -> bf16 ----------------------------
__global__ __launch_bounds__(256) void convert_bf16_kernel(
    const float* __restrict__ src, unsigned short* __restrict__ dst, int n4) {
  int i = blockIdx.x * 256 + threadIdx.x;
  if (i >= n4) return;
  float4 v = ((const float4*)src)[i];
  uint2 p;
  p.x = pack2bf(v.x, v.y);
  p.y = pack2bf(v.z, v.w);
  ((uint2*)dst)[i] = p;
}

// --------- build per-position row lists from ragged region_counts ----------
// nj[j]            = number of flat rows whose position == j
// rowsIdx[j][slot] = flat row index of slot-th such row (ordered by sample b)
__global__ __launch_bounds__(512) void prep_index_kernel(
    const int* __restrict__ counts, int* __restrict__ nj,
    int* __restrict__ rowsIdx) {
  __shared__ int sc[BSAMP];
  const int t = threadIdx.x;           // one thread per sample b
  sc[t] = counts[t];
  __syncthreads();
  int pre = 0;                          // exclusive prefix sum of counts
  for (int b = 0; b < t; ++b) pre += sc[b];
  if (t < RP) {
    int c = 0;
    for (int b = 0; b < BSAMP; ++b) c += (sc[b] > t) ? 1 : 0;
    nj[t] = c;
  }
  const int myc = sc[t];
  for (int jj = 0; jj < myc; ++jj) {
    int slot = 0;                       // rank of sample t among counts[b']>jj
    for (int b = 0; b < t; ++b) slot += (sc[b] > jj) ? 1 : 0;
    rowsIdx[jj * BSAMP + slot] = pre + jj;
  }
}

// ------------------------------ main kernel --------------------------------
// grid = (row tiles of 16, RP positions), block = 256 threads (8 wave32)
__global__ __launch_bounds__(256) void se_wmma_kernel(
    const float* __restrict__ X,            // [N, 2048] f32
    const float* __restrict__ b1,           // [37, 128]
    const float* __restrict__ b2,           // [37, 2048]
    const unsigned short* __restrict__ W1bf,// [37, 128, 2048] bf16
    const unsigned short* __restrict__ W2bf,// [37, 2048, 128] bf16
    const int* __restrict__ nj,             // [37]
    const int* __restrict__ rowsIdx,        // [37, 512]
    float* __restrict__ out)                // [N, 2048] f32
{
  constexpr int LDA = CH + 8;   // bf16 stride; 4112B row -> bank stride 4
  constexpr int LDH = HID + 8;  // 272B row -> bank stride 4
  __shared__ unsigned short ldsA[16 * LDA]; // 65792 B: 16-row x-tile (bf16)
  __shared__ unsigned short ldsH[16 * LDH]; //  4352 B: hidden tile (bf16)

  const int j    = blockIdx.y;
  const int tile = blockIdx.x;
  const int njv  = nj[j];
  if (tile * 16 >= njv) return;           // uniform across block: EXEC stays full

  const int tid  = threadIdx.x;
  const int wave = tid >> 5;
  const int lane = tid & 31;

  // Phase 1: gather 16 rows of x, convert f32 -> bf16 into LDS
  {
    const int m    = tid >> 4;                 // row within tile (0..15)
    const int cseg = (tid & 15) * (CH / 16);   // 128-col segment
    int idx = tile * 16 + m;
    if (idx > njv - 1) idx = njv - 1;          // clamp (stores predicated later)
    const int r = rowsIdx[j * BSAMP + idx];
    const float4* src = (const float4*)(X + (size_t)r * CH + cseg);
    unsigned short* dst = &ldsA[m * LDA + cseg];
#pragma unroll
    for (int q = 0; q < (CH / 16) / 4; ++q) {  // 32 x float4
      float4 v = src[q];
      uint2 p;
      p.x = pack2bf(v.x, v.y);
      p.y = pack2bf(v.z, v.w);
      *(uint2*)(dst + q * 4) = p;
    }
  }
  __syncthreads();

  // WMMA fragment addressing (CDNA5 bf16 layouts, wave32):
  //  A (16x32): lane m = lane%16 holds K seg0 = kk+(lane/16)*8 .. +7,
  //             seg1 = seg0+16 .. +23
  //  B (32x16): lane n = lane%16 holds K seg0 = kk+(lane/16)*8 .. +7, seg1 = seg0+16 .. +23
  //  D (16x16): vgpr i, lane -> (m = i + 8*(lane/16), n = lane%16)
  const int mA = lane & 15;
  const int s0 = (lane >> 4) * 8;
  const int kb = (lane >> 4) * 8;

  // Phase 2: GEMM1  H[16x128] = Xtile[16x2048] * W1[j]^T, wave w owns 16 hiddens
  {
    v8f_t acc = {};
    const unsigned short* wb =
        W1bf + ((size_t)j * HID + wave * 16 + (lane & 15)) * CH;
    for (int kk = 0; kk < CH; kk += 32) {
      Frag16 a, b;
      a.u[0] = *(const uint4*)&ldsA[mA * LDA + kk + s0];
      a.u[1] = *(const uint4*)&ldsA[mA * LDA + kk + 16 + s0];
      b.u[0] = *(const uint4*)(wb + kk + kb);
      b.u[1] = *(const uint4*)(wb + kk + kb + 16);
      acc = __builtin_amdgcn_wmma_f32_16x16x32_bf16(false, a.v, false, b.v,
                                                    (short)0, acc, false, false);
    }
    const float bias = b1[j * HID + wave * 16 + (lane & 15)];
    const int mBase = (lane >> 4) * 8;
#pragma unroll
    for (int i = 0; i < 8; ++i) {
      float h = acc[i] + bias;
      h = h > 0.f ? h : 0.f;                   // relu
      ldsH[(mBase + i) * LDH + wave * 16 + (lane & 15)] = f2bf(h);
    }
  }
  __syncthreads();

  // Per-vgpr output rows (fixed across column tiles)
  const int mBase = (lane >> 4) * 8;
  int  rRow[8];
  bool vRow[8];
#pragma unroll
  for (int i = 0; i < 8; ++i) {
    int idx = tile * 16 + mBase + i;
    vRow[i] = (idx < njv);
    if (idx > njv - 1) idx = njv - 1;
    rRow[i] = rowsIdx[j * BSAMP + idx];
  }

  // Phase 3: GEMM2  S[16x2048] = H[16x128] * W2[j]^T, streamed 16 cols at a time,
  // fused sigmoid + residual + scatter store
  for (int ct = wave; ct < CH / 16; ct += 8) {
    const int cCol = ct * 16 + (lane & 15);
    v8f_t acc = {};
    const unsigned short* wb = W2bf + ((size_t)j * CH + cCol) * HID;
#pragma unroll
    for (int kk = 0; kk < HID; kk += 32) {
      Frag16 a, b;
      a.u[0] = *(const uint4*)&ldsH[mA * LDH + kk + s0];
      a.u[1] = *(const uint4*)&ldsH[mA * LDH + kk + 16 + s0];
      b.u[0] = *(const uint4*)(wb + kk + kb);
      b.u[1] = *(const uint4*)(wb + kk + kb + 16);
      acc = __builtin_amdgcn_wmma_f32_16x16x32_bf16(false, a.v, false, b.v,
                                                    (short)0, acc, false, false);
    }
    const float bias = b2[(size_t)j * CH + cCol];
#pragma unroll
    for (int i = 0; i < 8; ++i) {
      if (vRow[i]) {
        float t = acc[i] + bias;
        float s = 1.0f / (1.0f + __expf(-t));  // sigmoid
        size_t off = (size_t)rRow[i] * CH + cCol;
        out[off] = X[off] * (1.0f + s);        // residual x*(1+s)
      }
    }
  }
}

// ------------------------------- launcher ----------------------------------
extern "C" void kernel_launch(void* const* d_in, const int* in_sizes, int n_in,
                              void* d_out, int out_size, void* d_ws, size_t ws_size,
                              hipStream_t stream) {
  (void)in_sizes; (void)n_in; (void)out_size; (void)ws_size;
  const float* X      = (const float*)d_in[0];   // feature_tens [N,2048]
  const int*   counts = (const int*)d_in[1];     // region_counts [512]
  const float* W1     = (const float*)d_in[2];   // [37,128,2048]
  const float* b1     = (const float*)d_in[3];   // [37,128]
  const float* W2     = (const float*)d_in[4];   // [37,2048,128]
  const float* b2     = (const float*)d_in[5];   // [37,2048]
  float* out = (float*)d_out;

  // workspace layout: W1bf | W2bf | nj[37] | rowsIdx[37*512]   (~38.9 MB)
  const size_t Welems = (size_t)RP * HID * CH;   // 9,699,328 per tensor
  unsigned short* w1bf = (unsigned short*)d_ws;
  unsigned short* w2bf = w1bf + Welems;
  int* nj      = (int*)(w2bf + Welems);
  int* rowsIdx = nj + RP;

  const int n4     = (int)(Welems / 4);
  const int cblks  = (n4 + 255) / 256;
  convert_bf16_kernel<<<cblks, 256, 0, stream>>>(W1, w1bf, n4);
  convert_bf16_kernel<<<cblks, 256, 0, stream>>>(W2, w2bf, n4);
  prep_index_kernel<<<1, BSAMP, 0, stream>>>(counts, nj, rowsIdx);

  dim3 grid((BSAMP + 15) / 16, RP);              // 32 row-tiles x 37 positions
  se_wmma_kernel<<<grid, 256, 0, stream>>>(X, b1, b2, w1bf, w2bf, nj, rowsIdx, out);
}
